// Attention_26465588478052
// MI455X (gfx1250) — matrix operand order repaired and checked by reading, not compiled, not run on hardware
//
#include <hip/hip_runtime.h>
#include <hip/hip_bf16.h>
#include <math.h>

#define B_     4
#define C_     256
#define NPIX   4096
#define HEADS_ 4
#define HD     32
#define INNER  128
#define EPS_   1e-12f

typedef __attribute__((ext_vector_type(8)))  _Float16 h8;
typedef __attribute__((ext_vector_type(16))) _Float16 v16h;
typedef __attribute__((ext_vector_type(8)))  float    v8f;
typedef __attribute__((ext_vector_type(4)))  int      i4;
typedef __attribute__((address_space(3)))    i4       i4_lds;

#if defined(__has_builtin)
#if __has_builtin(__builtin_amdgcn_global_load_async_to_lds_b128)
#define HAVE_ASYNC_LDS 1
#endif
#endif
#ifndef HAVE_ASYNC_LDS
#define HAVE_ASYNC_LDS 0
#endif

static __device__ __forceinline__ v16h cat16(h8 a, h8 b) {
    return __builtin_shufflevector(a, b, 0,1,2,3,4,5,6,7,8,9,10,11,12,13,14,15);
}

static __device__ __forceinline__ v8f wmma16(v16h a, v16h b, v8f c) {
    return __builtin_amdgcn_wmma_f32_16x16x32_f16(false, a, false, b, (short)0, c, false, false);
}

static __device__ __forceinline__ void wait_async0() {
#if HAVE_ASYNC_LDS
#if __has_builtin(__builtin_amdgcn_s_wait_asynccnt)
    __builtin_amdgcn_s_wait_asynccnt(0);
#else
    asm volatile("s_wait_asynccnt 0" ::: "memory");
#endif
#endif
}

// Stage one 32x32 f16 tile pair: K row-major (async DMA to LDS when available)
// and V transposed (for B-fragment column access).
static __device__ __forceinline__ void stage_tiles(
    const _Float16* kp, const _Float16* vp,
    _Float16* kbuf, _Float16* vbuf, int srow, int sseg)
{
#if HAVE_ASYNC_LDS
    __builtin_amdgcn_global_load_async_to_lds_b128(
        (i4*)kp, (i4_lds*)&kbuf[srow * 40 + sseg * 8], 0, 0);
#else
    *(h8*)&kbuf[srow * 40 + sseg * 8] = *(const h8*)kp;
#endif
    h8 vv = *(const h8*)vp;
#pragma unroll
    for (int i = 0; i < 8; i++) vbuf[(sseg * 8 + i) * 40 + srow] = vv[i];
}

// ---------------------------------------------------------------------------
// Kernel 0: one-time weight conversion f32 -> f16 (weights are L2-resident).
// ---------------------------------------------------------------------------
__global__ __launch_bounds__(256) void k_cvt_w(
    const float* __restrict__ wqkv, const float* __restrict__ wout,
    _Float16* __restrict__ wqkv_h, _Float16* __restrict__ wout_h)
{
    int i = blockIdx.x * 256 + threadIdx.x;
    if (i < 384 * C_)  wqkv_h[i] = (_Float16)wqkv[i];
    if (i < C_ * INNER) wout_h[i] = (_Float16)wout[i];
}

// ---------------------------------------------------------------------------
// Kernel 1: RMSNorm + QKV projection.  Tile = 16 pixels x 384 outputs.
// ---------------------------------------------------------------------------
__global__ __launch_bounds__(256) void k_rmsnorm_qkv(
    const float* __restrict__ x, const float* __restrict__ g,
    const _Float16* __restrict__ wqkv_h,
    _Float16* __restrict__ q, _Float16* __restrict__ k, _Float16* __restrict__ v)
{
    __shared__ _Float16 xn[16 * 264];   // [pixel][channel], padded row stride
    __shared__ float    ssum[16];

    const int b  = blockIdx.y;
    const int p0 = blockIdx.x * 16;
    const int t  = threadIdx.x;
    const int pl = t & 15;   // pixel local
    const int cg = t >> 4;   // channel group

    if (t < 16) ssum[t] = 0.0f;
    __syncthreads();

    const float* xb = x + (size_t)b * C_ * NPIX + p0;
    float xv[16];
    float acc = 0.0f;
#pragma unroll
    for (int kk = 0; kk < 16; kk++) {
        int c = cg * 16 + kk;
        float val = xb[(size_t)c * NPIX + pl];
        xv[kk] = val;
        acc += val * val;
    }
    atomicAdd(&ssum[pl], acc);
    __syncthreads();

    float s = rsqrtf(ssum[pl] * (1.0f / C_) + EPS_);
#pragma unroll
    for (int kk = 0; kk < 16; kk++) {
        int c = cg * 16 + kk;
        xn[pl * 264 + c] = (_Float16)(xv[kk] * s * g[c]);
    }
    __syncthreads();

    const int wave   = t >> 5;
    const int lane   = t & 31;
    const int nl     = lane & 15;
    const int hi     = lane >> 4;
    const int off_lo = hi * 8;

    // hoist the M-tile-invariant activation B-fragments (all 8 K-blocks)
    v16h bfr[8];
#pragma unroll
    for (int kbi = 0; kbi < 8; kbi++) {
        int kb = kbi * 32;
        bfr[kbi] = cat16(*(const h8*)&xn[nl * 264 + kb + hi * 8],
                         *(const h8*)&xn[nl * 264 + kb + hi * 8 + 16]);
    }

#pragma unroll
    for (int mt = 0; mt < 3; mt++) {
        const int m0 = (wave * 3 + mt) * 16;
        v8f c8 = {};
#pragma unroll
        for (int kbi = 0; kbi < 8; kbi++) {
            const _Float16* wr = wqkv_h + (size_t)(m0 + nl) * C_ + kbi * 32;
            v16h a = cat16(*(const h8*)(wr + off_lo), *(const h8*)(wr + off_lo + 16));
            c8 = wmma16(a, bfr[kbi], c8);
        }
        const int pix = p0 + nl;
#pragma unroll
        for (int j = 0; j < 8; j++) {
            int o   = m0 + j + hi * 8;       // output channel (0..383)
            int sel = o >> 7;                // 0=q 1=k 2=v
            int hh  = (o >> 5) & 3;
            int d   = o & 31;
            _Float16* dst = (sel == 0) ? q : ((sel == 1) ? k : v);
            dst[(((size_t)b * HEADS_ + hh) * NPIX + pix) * HD + d] = (_Float16)c8[j];
        }
    }
}

// ---------------------------------------------------------------------------
// Kernel 2: flash attention, transposed-score formulation with double-buffered
// async K/V staging.  simT = K_tile x Q^T : its D layout IS the A layout of P
// for P x V, so no LDS transpose of P is needed.  4 waves x 16 query rows.
// ---------------------------------------------------------------------------
__global__ __launch_bounds__(128) void k_attn(
    const _Float16* __restrict__ q, const _Float16* __restrict__ k,
    const _Float16* __restrict__ v, _Float16* __restrict__ attn_out)
{
    __shared__ _Float16 klds[2][32 * 40];   // K tiles, row-major [j][d]
    __shared__ _Float16 vldsT[2][32 * 40];  // V tiles transposed [d][j]

    const int bh   = blockIdx.y;
    const int qt   = blockIdx.x;
    const int t    = threadIdx.x;
    const int wave = t >> 5;
    const int lane = t & 31;
    const int nl     = lane & 15;
    const int hi     = lane >> 4;
    const int off_lo = hi * 8;

    const int    qrow0 = qt * 64 + wave * 16;
    const size_t base  = (size_t)bh * NPIX * HD;

    // Q as persistent B-fragment: lane = query col, K = d split by lane half
    const _Float16* qp = q + base + (size_t)(qrow0 + nl) * HD + hi * 8;
    v16h qb = cat16(*(const h8*)qp, *(const h8*)(qp + 16));

    float m_run = -1e30f, l_run = 0.0f;   // per-lane (= per-query) softmax state
    v8f acc0 = {}, acc1 = {};
    const float scale = 0.17677669529663687f;  // 1/sqrt(32)

    const int srow = t >> 2;  // 0..31 staging row
    const int sseg = t & 3;
    const int soff = srow * HD + sseg * 8;
    const _Float16* kbase = k + base;
    const _Float16* vbase = v + base;

    // prologue: stage tile 0
    stage_tiles(kbase + soff, vbase + soff, klds[0], vldsT[0], srow, sseg);
    wait_async0();
    __syncthreads();

#pragma unroll 1
    for (int j0 = 0; j0 < NPIX; j0 += 32) {
        const int cur = (j0 >> 5) & 1;
        const int jn  = j0 + 32;
        if (jn < NPIX) {   // block-uniform: EXEC stays all-ones at the WMMAs
            stage_tiles(kbase + jn * HD + soff, vbase + jn * HD + soff,
                        klds[cur ^ 1], vldsT[cur ^ 1], srow, sseg);
            __builtin_prefetch(kbase + (j0 + 64) * HD + soff, 0, 1);
            __builtin_prefetch(vbase + (j0 + 64) * HD + soff, 0, 1);
        }
        const _Float16* kc = klds[cur];
        const _Float16* vc = vldsT[cur];

        // ---- simT = K_rows * Q^T (two 16-row key sub-tiles) ----
        v16h ka0 = cat16(*(const h8*)&kc[nl * 40 + off_lo],
                         *(const h8*)&kc[nl * 40 + off_lo + 16]);
        v16h ka1 = cat16(*(const h8*)&kc[(nl + 16) * 40 + off_lo],
                         *(const h8*)&kc[(nl + 16) * 40 + off_lo + 16]);
        v8f s0 = {}, s1 = {};
        s0 = wmma16(ka0, qb, s0);   // D: lane = query, vgpr = key row (j 0..15)
        s1 = wmma16(ka1, qb, s1);   // j 16..31

        // ---- per-lane online softmax over this lane's 16 of the 32 keys ----
        float p0[8], p1[8];
        float mx = -1e30f;
#pragma unroll
        for (int jv = 0; jv < 8; jv++) {
            p0[jv] = s0[jv] * scale;
            p1[jv] = s1[jv] * scale;
            mx = fmaxf(mx, fmaxf(p0[jv], p1[jv]));
        }
        mx = fmaxf(mx, __shfl_xor(mx, 16, 32));   // combine lane-half partner
        float mnew = fmaxf(m_run, mx);
        float f    = __expf(m_run - mnew);
        float rs   = 0.0f;
#pragma unroll
        for (int jv = 0; jv < 8; jv++) {
            p0[jv] = __expf(p0[jv] - mnew);
            p1[jv] = __expf(p1[jv] - mnew);
            rs += p0[jv] + p1[jv];
        }
        rs += __shfl_xor(rs, 16, 32);
        l_run = l_run * f + rs;
        m_run = mnew;

        // ---- P is already in A-fragment layout: just narrow to f16 ----
        h8 pa0, pa1;
#pragma unroll
        for (int jv = 0; jv < 8; jv++) {
            pa0[jv] = (_Float16)p0[jv];
            pa1[jv] = (_Float16)p1[jv];
        }
        v16h pa = cat16(pa0, pa1);

        // ---- rescale accumulators with row-broadcast correction ----
#pragma unroll
        for (int jv = 0; jv < 8; jv++) {
            float fb = __shfl(f, jv + hi * 8, 32);
            acc0[jv] *= fb;
            acc1[jv] *= fb;
        }

        // ---- out += P * V ----
        v16h vb0 = cat16(*(const h8*)&vc[nl * 40 + hi * 8],
                         *(const h8*)&vc[nl * 40 + hi * 8 + 16]);
        v16h vb1 = cat16(*(const h8*)&vc[(nl + 16) * 40 + hi * 8],
                         *(const h8*)&vc[(nl + 16) * 40 + hi * 8 + 16]);
        acc0 = wmma16(pa, vb0, acc0);
        acc1 = wmma16(pa, vb1, acc1);

        wait_async0();       // next tile's DMA has overlapped the compute above
        __syncthreads();
    }

    // ---- normalize and emit [b, n, h*32+d] f16 for the projection GEMM ----
    const float linv = 1.0f / l_run;
    const int b = bh >> 2, h = bh & 3;
#pragma unroll
    for (int jv = 0; jv < 8; jv++) {
        float lb = __shfl(linv, jv + hi * 8, 32);
        int   rg = qrow0 + jv + hi * 8;
        attn_out[((size_t)b * NPIX + rg) * INNER + h * HD + nl]      = (_Float16)(acc0[jv] * lb);
        attn_out[((size_t)b * NPIX + rg) * INNER + h * HD + 16 + nl] = (_Float16)(acc1[jv] * lb);
    }
}

// ---------------------------------------------------------------------------
// Kernel 3: output projection + bias.  Tile = 16 pixels x 256 outputs.
// ---------------------------------------------------------------------------
__global__ __launch_bounds__(256) void k_proj(
    const _Float16* __restrict__ attn, const _Float16* __restrict__ wout_h,
    const float* __restrict__ bout, float* __restrict__ out)
{
    const int b  = blockIdx.y;
    const int p0 = blockIdx.x * 16;
    const int t  = threadIdx.x;
    const int wave   = t >> 5;
    const int lane   = t & 31;
    const int nl     = lane & 15;
    const int hi     = lane >> 4;
    const int off_lo = hi * 8;

    v16h bf[4];
    const _Float16* ap = attn + ((size_t)b * NPIX + p0 + nl) * INNER;
#pragma unroll
    for (int kbi = 0; kbi < 4; kbi++) {
        int kb = kbi * 32;
        bf[kbi] = cat16(*(const h8*)(ap + kb + hi * 8),
                        *(const h8*)(ap + kb + hi * 8 + 16));
    }

#pragma unroll
    for (int mt = 0; mt < 2; mt++) {
        const int m0 = (wave * 2 + mt) * 16;
        v8f c8 = {};
#pragma unroll
        for (int kbi = 0; kbi < 4; kbi++) {
            const _Float16* wr = wout_h + (size_t)(m0 + nl) * INNER + kbi * 32;
            c8 = wmma16(cat16(*(const h8*)(wr + off_lo), *(const h8*)(wr + off_lo + 16)),
                        bf[kbi], c8);
        }
#pragma unroll
        for (int j = 0; j < 8; j++) {
            int o = m0 + j + hi * 8;
            out[((size_t)b * C_ + o) * NPIX + p0 + nl] = c8[j] + bout[o];
        }
    }
}

// ---------------------------------------------------------------------------
extern "C" void kernel_launch(void* const* d_in, const int* in_sizes, int n_in,
                              void* d_out, int out_size, void* d_ws, size_t ws_size,
                              hipStream_t stream)
{
    (void)in_sizes; (void)n_in; (void)out_size; (void)ws_size;
    const float* x    = (const float*)d_in[0];
    const float* g    = (const float*)d_in[1];
    const float* wqkv = (const float*)d_in[2];
    const float* wout = (const float*)d_in[3];
    const float* bout = (const float*)d_in[4];
    float*       out  = (float*)d_out;

    _Float16* ws = (_Float16*)d_ws;
    const size_t qkv_sz = (size_t)B_ * HEADS_ * NPIX * HD;  // 2M halfs each
    _Float16* q      = ws;
    _Float16* k      = q + qkv_sz;
    _Float16* v      = k + qkv_sz;
    _Float16* attn   = v + qkv_sz;                 // B_*NPIX*INNER halfs
    _Float16* wqkv_h = attn + (size_t)B_ * NPIX * INNER;
    _Float16* wout_h = wqkv_h + 384 * C_;

    k_cvt_w<<<dim3(384), dim3(256), 0, stream>>>(wqkv, wout, wqkv_h, wout_h);

    dim3 grd1(NPIX / 16, B_);
    k_rmsnorm_qkv<<<grd1, dim3(256), 0, stream>>>(x, g, wqkv_h, q, k, v);

    dim3 grd2(NPIX / 64, B_ * HEADS_);
    k_attn<<<grd2, dim3(128), 0, stream>>>(q, k, v, attn);

    dim3 grd3(NPIX / 16, B_);
    k_proj<<<grd3, dim3(256), 0, stream>>>(attn, wout_h, bout, out);
}
